// AttentionModule_67568425501573
// MI455X (gfx1250) — matrix operand order repaired and checked by reading, not compiled, not run on hardware
//
#include <hip/hip_runtime.h>
#include <math.h>
#include <stdint.h>

#define D_MODEL 1024
#define NH 16
#define HD 64
#define BATCH 2
#define SEQ 2048
#define M_TOTAL (BATCH * SEQ)   // 4096
#define N_QKV (3 * D_MODEL)     // 3072

typedef __attribute__((ext_vector_type(16))) __bf16 v16bf;
typedef __attribute__((ext_vector_type(8)))  float  v8f;

__device__ __forceinline__ __bf16 f2bf(float f) {
  union { float f; unsigned u; } a; a.f = f;
  unsigned u = a.u + 0x7FFFu + ((a.u >> 16) & 1u);  // round-to-nearest-even
  union { unsigned short s; __bf16 b; } r; r.s = (unsigned short)(u >> 16);
  return r.b;
}

// CDNA5 async tile staging: global -> LDS without VGPR transit (ASYNCcnt).
__device__ __forceinline__ void async_load_b128x2(const void* gptr, void* ldsptr) {
  unsigned l = (unsigned)(uintptr_t)ldsptr;
  unsigned long long g = (unsigned long long)(uintptr_t)gptr;
  asm volatile("global_load_async_to_lds_b128 %0, %1, off"
               :: "v"(l), "v"(g) : "memory");
  asm volatile("global_load_async_to_lds_b128 %0, %1, off offset:16"
               :: "v"(l), "v"(g) : "memory");
}
__device__ __forceinline__ void wait_asynccnt0() {
  asm volatile("s_wait_asynccnt 0x0" ::: "memory");
}

// ---------------------------------------------------------------------------
// Kernel 1: fp32 -> bf16 conversions (+ weight transposes to [K][N] layout)
// ---------------------------------------------------------------------------
__global__ __launch_bounds__(256)
void convert_kernel(const float* __restrict__ x, const float* __restrict__ qkvw,
                    const float* __restrict__ fcw,
                    __bf16* __restrict__ xb, __bf16* __restrict__ wT,
                    __bf16* __restrict__ fT) {
  const size_t NX = (size_t)M_TOTAL * D_MODEL;   // 4,194,304
  const size_t NW = (size_t)N_QKV * D_MODEL;     // 3,145,728
  const size_t NF = (size_t)D_MODEL * D_MODEL;   // 1,048,576
  size_t i = (size_t)blockIdx.x * 256 + threadIdx.x;
  if (i < NX) {
    xb[i] = f2bf(x[i]);
  } else if (i < NX + NW) {
    size_t j = i - NX;
    size_t d = j / N_QKV, e = j % N_QKV;          // wT[d][e] = qkv_w[e][d]
    wT[j] = f2bf(qkvw[e * D_MODEL + d]);
  } else if (i < NX + NW + NF) {
    size_t j = i - NX - NW;
    size_t d = j / D_MODEL, e = j % D_MODEL;      // fT[d][e] = fc_w[e][d]
    fT[j] = f2bf(fcw[e * D_MODEL + d]);
  }
}

// ---------------------------------------------------------------------------
// Kernel 2/4: bf16 GEMM  C[M,N] = A[M,K] * B[K,N]  (B given as [K][N] bf16)
// Workgroup tile 128x128, 8 waves (4m x 2n), each wave 32x64 -> 2x4 wmma
// tiles => 8 v_wmma_f32_16x16x32_bf16 per K-step per wave.
// A tile staged via global_load_async_to_lds_b128 (no VGPR transit).
// mode 0: qkv epilogue (bias + positional encoding, scatter to q/k/v bf16)
// mode 1: fc epilogue (bias, fp32 store to d_out)
// ---------------------------------------------------------------------------
__global__ __launch_bounds__(256)
void gemm_bf16_kernel(const __bf16* __restrict__ A, const __bf16* __restrict__ Bm,
                      int M, int N, int K, int mode,
                      const float* __restrict__ bias,
                      __bf16* __restrict__ qout, __bf16* __restrict__ kout,
                      __bf16* __restrict__ vout, float* __restrict__ fout) {
  __shared__ __align__(16) __bf16 ldsA[128][32];   // [m][k]
  __shared__ __align__(16) __bf16 ldsB[128][32];   // [n][k]  (column-contiguous B)
  const int tid  = threadIdx.x;
  const int lane = tid & 31;
  const int wave = tid >> 5;
  const int hlf  = lane >> 4;     // half-wave select (ISA 16-bit operand layout)
  const int l16  = lane & 15;
  const int m0 = blockIdx.x * 128;
  const int n0 = blockIdx.y * 128;
  const int wm = (wave >> 1) * 32;
  const int wn = (wave & 1) * 64;

  v8f acc[2][4] = {};
  const int arow = tid >> 1;            // A staging: 128 rows x 32 cols
  const int acol = (tid & 1) * 16;
  const int bk   = tid >> 3;            // B staging: 32 k x 128 n (transposed write)
  const int bn   = (tid & 7) * 16;

  for (int k0 = 0; k0 < K; k0 += 32) {
    // stage A tile: async DMA straight into LDS (2 x 16B per thread)
    async_load_b128x2(A + (size_t)(m0 + arow) * K + k0 + acol, &ldsA[arow][acol]);
    // stage B tile transposed into [n][k] (element shuffle -> register transit)
    const uint4* gb = (const uint4*)(Bm + (size_t)(k0 + bk) * N + n0 + bn);
    uint4 b0 = gb[0], b1 = gb[1];
    const __bf16* be = (const __bf16*)&b0;
#pragma unroll
    for (int i = 0; i < 8; ++i) ldsB[bn + i][bk] = be[i];
    be = (const __bf16*)&b1;
#pragma unroll
    for (int i = 0; i < 8; ++i) ldsB[bn + 8 + i][bk] = be[i];
    wait_asynccnt0();
    __syncthreads();

    if (k0 + 32 < K) {  // pull next K tile toward the WGP (global_prefetch_b8)
      __builtin_prefetch(A + (size_t)(m0 + arow) * K + k0 + 32 + acol, 0, 3);
      __builtin_prefetch(Bm + (size_t)(k0 + 32 + bk) * N + n0 + bn, 0, 3);
    }

    union Fr { v16bf v; uint4 q4[2]; } fa[2], fb[4];
#pragma unroll
    for (int t = 0; t < 2; ++t) {       // A: lane m=l16, K groups split by half
      int row = wm + t * 16 + l16;
      fa[t].q4[0] = *(const uint4*)&ldsA[row][8 * hlf];
      fa[t].q4[1] = *(const uint4*)&ldsA[row][16 + 8 * hlf];
    }
#pragma unroll
    for (int t = 0; t < 4; ++t) {       // B: lane n=l16, K groups split by half
      int col = wn + t * 16 + l16;
      fb[t].q4[0] = *(const uint4*)&ldsB[col][8 * hlf];
      fb[t].q4[1] = *(const uint4*)&ldsB[col][16 + 8 * hlf];
    }
#pragma unroll
    for (int ti = 0; ti < 2; ++ti)
#pragma unroll
      for (int tj = 0; tj < 4; ++tj)
        acc[ti][tj] = __builtin_amdgcn_wmma_f32_16x16x32_bf16(
            false, fa[ti].v, false, fb[tj].v, (short)0, acc[ti][tj], false, false);
    __syncthreads();
  }

  // epilogue: C layout -> m = r + 8*hlf, n = l16 (per 16x16 tile)
#pragma unroll
  for (int ti = 0; ti < 2; ++ti) {
#pragma unroll
    for (int tj = 0; tj < 4; ++tj) {
#pragma unroll
      for (int r = 0; r < 8; ++r) {
        int m = m0 + wm + ti * 16 + r + 8 * hlf;
        int n = n0 + wn + tj * 16 + l16;
        float val = acc[ti][tj][r] + bias[n];
        if (mode == 0) {
          int part = n >> 10;          // 0=q 1=k 2=v
          int dd = n & 1023;
          int b = m >> 11;             // row = b*SEQ + s
          int s = m & 2047;
          if (part < 2) {
            // reference quirk: pe = make_pe(B, D)[:, None, :] -> one row per batch
            float div = __expf((float)(dd & ~1) * (-9.210340371976184f / 1024.0f));
            float ang = (float)b * div;
            val += (dd & 1) ? __cosf(ang) : __sinf(ang);
          }
          int h = dd >> 6, hd = dd & 63;
          size_t didx = (((size_t)(b * NH + h) * SEQ + s) << 6) + hd;
          __bf16* dst = (part == 0) ? qout : (part == 1) ? kout : vout;
          dst[didx] = f2bf(val);
        } else {
          fout[(size_t)m * N + n] = val;
        }
      }
    }
  }
}

// ---------------------------------------------------------------------------
// Kernel 3: flash attention. grid = (SEQ/128, B*H). 8 waves x 16 q-rows each.
// Streams 64-key tiles: S=Q*K^T (wmma) -> scale/mask -> online softmax (fp32)
// -> P through per-wave LDS (C->A relayout) -> O += P*V (wmma) with rescale.
// K tile staged via async-to-LDS; V tile needs a transpose so stays manual.
// ---------------------------------------------------------------------------
__global__ __launch_bounds__(256)
void flash_attn_kernel(const __bf16* __restrict__ q, const __bf16* __restrict__ k,
                       const __bf16* __restrict__ v, const int* __restrict__ mask,
                       __bf16* __restrict__ out) {
  __shared__ __align__(16) __bf16 ldsK[64][64];      // [kpos][hd] == B-op [n][k]
  __shared__ __align__(16) __bf16 ldsV[64][64];      // [hd][kpos] (transposed)
  __shared__ __align__(16) __bf16 ldsP[8][16][64];   // per-wave P relayout slab
  const int tid  = threadIdx.x;
  const int lane = tid & 31;
  const int wave = tid >> 5;
  const int hlf  = lane >> 4;
  const int l16  = lane & 15;
  const int bh = blockIdx.y;                 // b*NH + h
  const int b = bh >> 4, h = bh & 15;
  const int q0 = blockIdx.x * 128 + wave * 16;
  const size_t headbase = (size_t)bh * SEQ * HD;

  // Q rows live in registers as WMMA A fragments for the whole k-loop
  union Fr { v16bf v; uint4 q4[2]; } fq[2];
  {
    const __bf16* qrow = q + headbase + (size_t)(q0 + l16) * HD;
#pragma unroll
    for (int ks = 0; ks < 2; ++ks) {
      fq[ks].q4[0] = *(const uint4*)(qrow + ks * 32 + 8 * hlf);
      fq[ks].q4[1] = *(const uint4*)(qrow + ks * 32 + 16 + 8 * hlf);
    }
  }

  float rmax[8], rsum[8];
#pragma unroll
  for (int r = 0; r < 8; ++r) { rmax[r] = -3.0e38f; rsum[r] = 0.0f; }
  v8f o[4] = {};

  const int krow = tid >> 2;        // staging: 64 rows x 64 cols
  const int kcb  = (tid & 3) * 16;

  for (int kb = 0; kb < SEQ; kb += 64) {
    __syncthreads();  // previous iteration done reading ldsK/ldsV
    // K tile: async DMA straight into LDS (row-major == B-operand layout)
    async_load_b128x2(k + headbase + (size_t)(kb + krow) * HD + kcb,
                      &ldsK[krow][kcb]);
    // V tile: transposed store (register transit)
    const uint4* gv = (const uint4*)(v + headbase + (size_t)(kb + krow) * HD + kcb);
    uint4 v0 = gv[0], v1 = gv[1];
    const __bf16* ve = (const __bf16*)&v0;
#pragma unroll
    for (int i = 0; i < 8; ++i) ldsV[kcb + i][krow] = ve[i];
    ve = (const __bf16*)&v1;
#pragma unroll
    for (int i = 0; i < 8; ++i) ldsV[kcb + 8 + i][krow] = ve[i];
    wait_asynccnt0();
    __syncthreads();

    // S = Q * K^T : 4 subtiles of 16 key positions
    v8f s[4];
#pragma unroll
    for (int nt = 0; nt < 4; ++nt) {
      v8f accs = {};
#pragma unroll
      for (int ks = 0; ks < 2; ++ks) {
        union Fr fb;
        int col = nt * 16 + l16;
        fb.q4[0] = *(const uint4*)&ldsK[col][ks * 32 + 8 * hlf];
        fb.q4[1] = *(const uint4*)&ldsK[col][ks * 32 + 16 + 8 * hlf];
        accs = __builtin_amdgcn_wmma_f32_16x16x32_bf16(
            false, fq[ks].v, false, fb.v, (short)0, accs, false, false);
      }
      s[nt] = accs;
    }
    // scale by 1/sqrt(64), apply mask (mask==1 -> -1e9, replacing the score)
#pragma unroll
    for (int nt = 0; nt < 4; ++nt) {
      int kc = kb + nt * 16 + l16;
#pragma unroll
      for (int r = 0; r < 8; ++r) {
        int qr = q0 + r + 8 * hlf;
        float sv = s[nt][r] * 0.125f;
        int mk = mask[(size_t)qr * SEQ + kc];
        s[nt][r] = (mk == 1) ? -1.0e9f : sv;
      }
    }
    // online softmax: rows of a C-tile live in one 16-lane half-group
#pragma unroll
    for (int r = 0; r < 8; ++r) {
      float mx = fmaxf(fmaxf(s[0][r], s[1][r]), fmaxf(s[2][r], s[3][r]));
#pragma unroll
      for (int off = 1; off < 16; off <<= 1)
        mx = fmaxf(mx, __shfl_xor(mx, off, 32));
      float nm = fmaxf(rmax[r], mx);
      float sc = __expf(rmax[r] - nm);
      rmax[r] = nm;
      float ts = 0.0f;
#pragma unroll
      for (int nt = 0; nt < 4; ++nt) {
        float p = __expf(s[nt][r] - nm);
        s[nt][r] = p;
        ts += p;
      }
#pragma unroll
      for (int off = 1; off < 16; off <<= 1)
        ts += __shfl_xor(ts, off, 32);
      rsum[r] = rsum[r] * sc + ts;
#pragma unroll
      for (int nt = 0; nt < 4; ++nt) o[nt][r] *= sc;
    }
    // P: C layout -> LDS -> A layout
#pragma unroll
    for (int nt = 0; nt < 4; ++nt)
#pragma unroll
      for (int r = 0; r < 8; ++r)
        ldsP[wave][r + 8 * hlf][nt * 16 + l16] = f2bf(s[nt][r]);
    __syncthreads();

    union Fr fp[2];
#pragma unroll
    for (int ks = 0; ks < 2; ++ks) {
      fp[ks].q4[0] = *(const uint4*)&ldsP[wave][l16][ks * 32 + 8 * hlf];
      fp[ks].q4[1] = *(const uint4*)&ldsP[wave][l16][ks * 32 + 16 + 8 * hlf];
    }
    // O += P * V
#pragma unroll
    for (int nt = 0; nt < 4; ++nt) {
#pragma unroll
      for (int ks = 0; ks < 2; ++ks) {
        union Fr fb;
        int col = nt * 16 + l16;
        fb.q4[0] = *(const uint4*)&ldsV[col][ks * 32 + 8 * hlf];
        fb.q4[1] = *(const uint4*)&ldsV[col][ks * 32 + 16 + 8 * hlf];
        o[nt] = __builtin_amdgcn_wmma_f32_16x16x32_bf16(
            false, fp[ks].v, false, fb.v, (short)0, o[nt], false, false);
      }
    }
  }
  // epilogue: normalize and pack back to [B, S, D] bf16 for the FC GEMM
#pragma unroll
  for (int nt = 0; nt < 4; ++nt) {
#pragma unroll
    for (int r = 0; r < 8; ++r) {
      int qr = q0 + r + 8 * hlf;
      int hd = nt * 16 + l16;
      float val = o[nt][r] / rsum[r];
      out[(size_t)(b * SEQ + qr) * D_MODEL + h * HD + hd] = f2bf(val);
    }
  }
}

// ---------------------------------------------------------------------------
extern "C" void kernel_launch(void* const* d_in, const int* in_sizes, int n_in,
                              void* d_out, int out_size, void* d_ws, size_t ws_size,
                              hipStream_t stream) {
  (void)in_sizes; (void)n_in; (void)out_size; (void)ws_size;
  const float* x    = (const float*)d_in[0];
  const int*   mask = (const int*)d_in[1];
  const float* qkvw = (const float*)d_in[2];
  const float* qkvb = (const float*)d_in[3];
  const float* fcw  = (const float*)d_in[4];
  const float* fcb  = (const float*)d_in[5];
  char* ws = (char*)d_ws;
  __bf16* xb = (__bf16*)(ws + 0);          //  8 MB x bf16
  __bf16* wT = (__bf16*)(ws + 8388608);    //  6 MB qkv_w^T bf16
  __bf16* fT = (__bf16*)(ws + 14680064);   //  2 MB fc_w^T bf16
  __bf16* qb = (__bf16*)(ws + 16777216);   //  8 MB q [B,H,S,Hd]
  __bf16* kb = (__bf16*)(ws + 25165824);   //  8 MB k
  __bf16* vb = (__bf16*)(ws + 33554432);   //  8 MB v
  __bf16* ao = (__bf16*)(ws + 41943040);   //  8 MB attn out [B,S,D]
  float* outp = (float*)d_out;

  convert_kernel<<<32768, 256, 0, stream>>>(x, qkvw, fcw, xb, wT, fT);
  gemm_bf16_kernel<<<dim3(32, 24), 256, 0, stream>>>(
      xb, wT, M_TOTAL, N_QKV, D_MODEL, 0, qkvb, qb, kb, vb, nullptr);
  flash_attn_kernel<<<dim3(16, 32), 256, 0, stream>>>(qb, kb, vb, mask, ao);
  gemm_bf16_kernel<<<dim3(32, 8), 256, 0, stream>>>(
      ao, fT, M_TOTAL, D_MODEL, D_MODEL, 1, fcb, nullptr, nullptr, nullptr, outp);
}
